// GRU_52097953300760
// MI455X (gfx1250) — compile-verified, matrix operand order checked by reading, not run on hardware
//
#include <hip/hip_runtime.h>
#include <hip/hip_bf16.h>

// Problem constants (from reference)
#define T_STEPS 8192
#define DIN     1024
#define H       1024
#define H3      3072

// Scan configuration: 64 persistent workgroups, each owns 16 hidden units
#define SCAN_BLOCKS 64
#define SLICE       16      // h-indices per block
#define COLS        48      // 3 gates * SLICE columns of Wh per block

typedef __bf16 bf16_t;
typedef bf16_t v16bf __attribute__((ext_vector_type(16)));
typedef bf16_t v8bf  __attribute__((ext_vector_type(8)));
typedef float  v8f   __attribute__((ext_vector_type(8)));

// ---- workspace layout (bytes) ----
// [0,256)                : grid-barrier counter
// [GI_OFF, +T*3H*4)      : gi_all fp32
// [A_OFF, +T*DIN*2)      : inputs converted to bf16
// [WT_OFF, +3H*DIN*2)    : Wi transposed (column-major -> rows) in bf16
#define GI_OFF  ((size_t)256)
#define A_OFF   (GI_OFF + (size_t)T_STEPS * H3 * 4)
#define WT_OFF  (A_OFF + (size_t)T_STEPS * DIN * 2)

// ---------------------------------------------------------------------------
__global__ void gru_init_ws(unsigned* cnt) {
    if (threadIdx.x == 0) *cnt = 0u;
}

// fp32 -> bf16 elementwise convert (inputs staging)
__global__ void gru_cvt_bf16(const float* __restrict__ in,
                             bf16_t* __restrict__ out, int n) {
    int i = blockIdx.x * blockDim.x + threadIdx.x;
    if (i < n) out[i] = (bf16_t)in[i];
}

// WiT[n][k] = Wi[k][n], fp32 -> bf16, tiled transpose through LDS
__global__ void gru_transpose_wi(const float* __restrict__ Wi,
                                 bf16_t* __restrict__ WiT) {
    __shared__ float tile[32][33];
    const int k0 = blockIdx.x * 32;   // K tile (rows of Wi)
    const int n0 = blockIdx.y * 32;   // N tile (cols of Wi)
    const int tx = threadIdx.x;       // 0..31
    const int ty = threadIdx.y;       // 0..7
    #pragma unroll
    for (int i = 0; i < 32; i += 8)
        tile[ty + i][tx] = Wi[(size_t)(k0 + ty + i) * H3 + (n0 + tx)];
    __syncthreads();
    #pragma unroll
    for (int i = 0; i < 32; i += 8)
        WiT[(size_t)(n0 + ty + i) * DIN + (k0 + tx)] = (bf16_t)tile[tx][ty + i];
}

// ---------------------------------------------------------------------------
// gi_all[t][c] = sum_k inputs[t][k]*Wi[k][c] + bi[c]
// One wave computes one 16x16 tile with v_wmma_f32_16x16x32_bf16, K-loop of 32.
//
// 16-bit A fragment layout (ISA 7.12.2): lane L (m = L&15):
//   lanes 0-15 : elems 0..7 -> K=0..7,  elems 8..15 -> K=16..23
//   lanes 16-31: elems 0..7 -> K=8..15, elems 8..15 -> K=24..31
// => per lane, two contiguous 8xbf16 (16B) runs at K = s and K = s+16,
//    s = (lane>=16)*8.  B uses the mirrored layout with n = L&15, fed from
//    the pre-transposed WiT so its runs are contiguous too.
__global__ void gru_gemm_wmma(const bf16_t* __restrict__ A,
                              const bf16_t* __restrict__ BT,
                              const float*  __restrict__ bi,
                              float* __restrict__ C) {
    const int lane = threadIdx.x & 31;
    const int wave = threadIdx.x >> 5;
    const int w    = blockIdx.x * 8 + wave;      // global wave-tile id
    const int tm   = w / (H3 / 16);              // 0..511
    const int tn   = w % (H3 / 16);              // 0..191
    const int half = lane >> 4;                  // 0 or 1
    const int mn   = lane & 15;

    const bf16_t* ap = A  + (size_t)(tm * 16 + mn) * DIN;  // A row (M = mn)
    const bf16_t* bp = BT + (size_t)(tn * 16 + mn) * DIN;  // Wi column (N = mn)

    v8f acc = {};
    for (int k0 = 0; k0 < DIN; k0 += 32) {
        const int kb = k0 + half * 8;
        v8bf alo = *(const v8bf*)(ap + kb);
        v8bf ahi = *(const v8bf*)(ap + kb + 16);
        v8bf blo = *(const v8bf*)(bp + kb);
        v8bf bhi = *(const v8bf*)(bp + kb + 16);
        v16bf av = __builtin_shufflevector(alo, ahi, 0,1,2,3,4,5,6,7,
                                           8,9,10,11,12,13,14,15);
        v16bf bv = __builtin_shufflevector(blo, bhi, 0,1,2,3,4,5,6,7,
                                           8,9,10,11,12,13,14,15);
        acc = __builtin_amdgcn_wmma_f32_16x16x32_bf16(
                  false, av, false, bv, (short)0, acc, false, false);
    }

    // D layout: VGPR r -> M = r + 8*half, N = lane&15
    const int col  = tn * 16 + mn;
    const float bb = bi[col];
    #pragma unroll
    for (int r = 0; r < 8; ++r) {
        const int row = tm * 16 + r + 8 * half;
        C[(size_t)row * H3 + col] = acc[r] + bb;
    }
}

// ---------------------------------------------------------------------------
// Persistent GRU scan. 64 blocks x 256 threads. Block b owns h[b*16 .. b*16+15]
// and Wh columns {j, H+j, 2H+j}. The 1024x48 f32 Wh slice lives in LDS for
// the whole scan (192 KB of the 320 KB/WGP). Grid sync = monotonic atomic.
__global__ void gru_scan(const float* __restrict__ gi,    // [T][3H]
                         const int*   __restrict__ term,  // [T]
                         const float* __restrict__ h0,    // [H]
                         const float* __restrict__ Wh,    // [H][3H]
                         const float* __restrict__ bhn,   // [H]
                         float* __restrict__ out,         // [2][T][H] (y, states)
                         unsigned* __restrict__ cnt) {
    extern __shared__ float smem[];
    float* whl = smem;                          // [1024][48]
    float* hsh = smem + (size_t)DIN * COLS;     // [1024]
    float* red = hsh + H;                       // [16 kg][16 jl][3]

    const int b   = blockIdx.x;
    const int tid = threadIdx.x;
    const int jl  = tid & 15;        // h-index within slice
    const int kg  = tid >> 4;        // K-chunk 0..15 (64 k's each)

    // Stage Wh slice: whl[k*48 + g*16 + j] = Wh[k][g*H + b*16 + j]
    for (int idx = tid; idx < DIN * COLS; idx += blockDim.x) {
        const int k = idx / COLS;
        const int c = idx % COLS;
        const int g = c >> 4, j = c & 15;
        whl[idx] = Wh[(size_t)k * H3 + g * H + b * SLICE + j];
    }
    float* states = out + (size_t)T_STEPS * H;   // second tuple element
    const int   jg = b * SLICE + jl;
    const float bh = bhn[jg];
    __syncthreads();

    for (int t = 0; t < T_STEPS; ++t) {
        // h_prev (with reset_on_terminate) into LDS
        const float* hprev = (t == 0) ? h0 : (states + (size_t)(t - 1) * H);
        const float  mask  = term[t] ? 0.0f : 1.0f;
        for (int k = tid; k < H; k += blockDim.x) hsh[k] = hprev[k] * mask;
        __syncthreads();

        // partial dots for the 3 gate columns of h-index jl over K chunk kg
        float p0 = 0.f, p1 = 0.f, p2 = 0.f;
        const float* wbase = whl + (size_t)(kg * 64) * COLS;
        const float* hbase = hsh + kg * 64;
        #pragma unroll 4
        for (int kk = 0; kk < 64; ++kk) {
            const float  hv = hbase[kk];
            const float* wr = wbase + (size_t)kk * COLS;
            p0 = fmaf(hv, wr[jl],        p0);
            p1 = fmaf(hv, wr[16 + jl],   p1);
            p2 = fmaf(hv, wr[32 + jl],   p2);
        }
        red[(kg * 16 + jl) * 3 + 0] = p0;
        red[(kg * 16 + jl) * 3 + 1] = p1;
        red[(kg * 16 + jl) * 3 + 2] = p2;
        __syncthreads();

        if (tid < 16) {   // jl == tid
            float ghr = 0.f, ghz = 0.f, ghn = 0.f;
            #pragma unroll
            for (int q = 0; q < 16; ++q) {
                ghr += red[(q * 16 + jl) * 3 + 0];
                ghz += red[(q * 16 + jl) * 3 + 1];
                ghn += red[(q * 16 + jl) * 3 + 2];
            }
            const float* gr = gi + (size_t)t * H3;
            const float r  = 1.f / (1.f + __expf(-(gr[jg]         + ghr)));
            const float z  = 1.f / (1.f + __expf(-(gr[H  + jg]    + ghz)));
            const float n  = tanhf(gr[2 * H + jg] + r * (ghn + bh));
            const float hp = hsh[jg];
            const float hn = (1.f - z) * n + z * hp;
            out[(size_t)t * H + jg]    = hn;   // y_t
            states[(size_t)t * H + jg] = hn;   // new_states (also next h)
        }
        __threadfence();     // release h_new before arriving
        __syncthreads();

        if (t < T_STEPS - 1) {
            if (tid == 0) {
                atomicAdd(cnt, 1u);
                const unsigned target = (unsigned)(t + 1) * SCAN_BLOCKS;
                while (__hip_atomic_load(cnt, __ATOMIC_ACQUIRE,
                                         __HIP_MEMORY_SCOPE_AGENT) < target)
                    __builtin_amdgcn_s_sleep(1);
            }
            __syncthreads();  // all threads see acquired state
        }
    }
}

// ---------------------------------------------------------------------------
extern "C" void kernel_launch(void* const* d_in, const int* in_sizes, int n_in,
                              void* d_out, int out_size, void* d_ws, size_t ws_size,
                              hipStream_t stream) {
    (void)in_sizes; (void)n_in; (void)out_size; (void)ws_size;
    const float* inputs = (const float*)d_in[0];
    const int*   term   = (const int*)  d_in[1];
    const float* h0     = (const float*)d_in[2];
    const float* Wi     = (const float*)d_in[3];
    const float* bi     = (const float*)d_in[4];
    const float* Wh     = (const float*)d_in[5];
    const float* bhn    = (const float*)d_in[6];
    float*       out    = (float*)d_out;

    char* ws = (char*)d_ws;
    unsigned* cnt = (unsigned*)ws;
    float*   gi   = (float*)  (ws + GI_OFF);
    bf16_t*  Abf  = (bf16_t*) (ws + A_OFF);
    bf16_t*  WiT  = (bf16_t*) (ws + WT_OFF);

    // 0) reset grid barrier
    gru_init_ws<<<1, 64, 0, stream>>>(cnt);

    // 1) inputs fp32 -> bf16
    {
        const int n = T_STEPS * DIN;
        gru_cvt_bf16<<<n / 256, 256, 0, stream>>>(inputs, Abf, n);
    }

    // 2) Wi -> WiT (bf16, column-contiguous for WMMA B fragments)
    gru_transpose_wi<<<dim3(DIN / 32, H3 / 32), dim3(32, 8), 0, stream>>>(Wi, WiT);

    // 3) gi_all = inputs @ Wi + bi  (WMMA bf16 -> f32)
    {
        const int tiles  = (T_STEPS / 16) * (H3 / 16);  // 98304 wave-tiles
        const int blocks = tiles / 8;                   // 8 waves/block
        gru_gemm_wmma<<<blocks, 256, 0, stream>>>(Abf, WiT, bi, gi);
    }

    // 4) persistent scan (Wh resident in LDS, atomic grid barrier per step)
    {
        const size_t smem = ((size_t)DIN * COLS + H + 16 * 16 * 3) * sizeof(float);
        (void)hipFuncSetAttribute((const void*)gru_scan,
                                  hipFuncAttributeMaxDynamicSharedMemorySize,
                                  (int)smem);
        gru_scan<<<SCAN_BLOCKS, 256, smem, stream>>>(gi, term, h0, Wh, bhn, out, cnt);
    }
}
